// Attention_13778255085613
// MI455X (gfx1250) — matrix operand order repaired and checked by reading, not compiled, not run on hardware
//
#include <hip/hip_runtime.h>
#include <math.h>

typedef __attribute__((ext_vector_type(16))) __bf16 v16bf;
typedef __attribute__((ext_vector_type(8)))  float  v8f;

#define H_       16
#define D_       64
#define DIM_     1024
#define N_       1024
#define TPAST_   4096
#define TK_      5120
#define MID_     3072
#define KEEPMID_ 2048
#define LDSP     72   // padded LDS row (ushorts): 144B, 16B-multiple, bank stride 36

// ---- f32 -> bf16 (round-to-nearest-even) helpers -------------------------
__device__ __forceinline__ unsigned bfr(float f) {
  union { float f; unsigned u; } c; c.f = f;
  return c.u + (0x7fffu + ((c.u >> 16) & 1u));
}
__device__ __forceinline__ unsigned pk2(float lo, float hi) {
#if __has_builtin(__builtin_amdgcn_cvt_pk_bf16_f32)
  return __builtin_bit_cast(unsigned, __builtin_amdgcn_cvt_pk_bf16_f32(lo, hi));
#else
  // one v_perm_b32 packs the two rounded high halves: {hi[31:16], lo[31:16]}
  return __builtin_amdgcn_perm(bfr(hi), bfr(lo), 0x07060302u);
#endif
}
__device__ __forceinline__ unsigned short f2bf(float f) {
  return (unsigned short)(bfr(f) >> 16);
}

union BfOp { v16bf v; uint4 q[2]; unsigned u[8]; unsigned short s[16]; };

// ---- async global->LDS copy (CDNA5 ASYNCcnt path), guarded ----------------
#if __has_builtin(__builtin_amdgcn_global_load_async_to_lds_b128) && \
    __has_builtin(__builtin_amdgcn_s_wait_asynccnt)
#define USE_ASYNC_LDS 1
typedef __attribute__((ext_vector_type(4))) int v4i_;
typedef __attribute__((address_space(1))) v4i_ gv4i;   // global AS int4
typedef __attribute__((address_space(3))) v4i_ lv4i;   // LDS AS int4
__device__ __forceinline__ void cp16_lds(const unsigned short* g, unsigned short* l) {
  __builtin_amdgcn_global_load_async_to_lds_b128(
      (gv4i*)(unsigned long long)(size_t)g,
      (lv4i*)(unsigned)(unsigned long long)(size_t)l,
      0, 0);
}
#else
#define USE_ASYNC_LDS 0
__device__ __forceinline__ void cp16_lds(const unsigned short* g, unsigned short* l) {
  *reinterpret_cast<uint4*>(l) = *reinterpret_cast<const uint4*>(g);
}
#endif

// ==========================================================================
// GEMM: C[M][Nc] = A[M][K] * W[Nc][K]^T + bias, bf16 WMMA, f32 accumulate.
// Block tile 64(M) x 64(N), K-slab 32. 128 threads = 4 waves; wave w owns
// cols [nbase+16w, +16), 4 M-subtiles -> 4 WMMAs per slab sharing one B op.
// mode 0: row-major store out0[M][Nc];  mode 1: QKV scatter to q/k/v [h][n][d]
// ==========================================================================
__global__ __launch_bounds__(128) void gemm_bf16_kernel(
    const float* __restrict__ A, const float* __restrict__ W,
    const float* __restrict__ bias, int M, int K, int Nc, int mode,
    float* __restrict__ out0, float* __restrict__ out1, float* __restrict__ out2)
{
  __shared__ __align__(16) unsigned short a_s[64][LDSP];
  __shared__ __align__(16) unsigned short b_s[64][LDSP];
  const int tid  = threadIdx.x;
  const int lane = tid & 31, wave = tid >> 5;
  const int nbase = blockIdx.x * 64;
  const int mbase = blockIdx.y * 64;
  v8f c0 = {}, c1 = {}, c2 = {}, c3 = {};
  const int nIter = K >> 5;
  for (int kb = 0; kb < nIter; ++kb) {
    const int k0 = kb << 5;
    // -- issue all staging loads first (overlap), then convert ------------
    float4 av[4], wv[4];
#pragma unroll
    for (int i = 0; i < 4; ++i) {
      const int fid = tid + (i << 7);
      const int row = fid >> 3, cc = (fid & 7) << 2;
      av[i] = *reinterpret_cast<const float4*>(&A[(size_t)(mbase + row) * K + k0 + cc]);
      wv[i] = *reinterpret_cast<const float4*>(&W[(size_t)(nbase + row) * K + k0 + cc]);
    }
    if (kb + 1 < nIter) {  // global_prefetch_b8 for next K-slab
      __builtin_prefetch(&A[(size_t)(mbase + (tid >> 1)) * K + k0 + 32], 0, 1);
      __builtin_prefetch(&W[(size_t)(nbase + (tid >> 1)) * K + k0 + 32], 0, 1);
    }
#pragma unroll
    for (int i = 0; i < 4; ++i) {
      const int fid = tid + (i << 7);
      const int row = fid >> 3, cc = (fid & 7) << 2;
      *reinterpret_cast<unsigned*>(&a_s[row][cc])     = pk2(av[i].x, av[i].y);
      *reinterpret_cast<unsigned*>(&a_s[row][cc + 2]) = pk2(av[i].z, av[i].w);
      *reinterpret_cast<unsigned*>(&b_s[row][cc])     = pk2(wv[i].x, wv[i].y);
      *reinterpret_cast<unsigned*>(&b_s[row][cc + 2]) = pk2(wv[i].z, wv[i].w);
    }
    __syncthreads();
    {
      const int m = lane & 15, hf = lane >> 4;
      // B 32x16 layout: lane(n,hf): e[0..7]=W^T[k0+8hf+j][col], e[8..15]=W^T[k0+16+8hf+j][col]
      BfOp b;
      const int colLoc = (wave << 4) + m;
      b.q[0] = *reinterpret_cast<const uint4*>(&b_s[colLoc][hf * 8]);
      b.q[1] = *reinterpret_cast<const uint4*>(&b_s[colLoc][16 + hf * 8]);
#pragma unroll
      for (int mt = 0; mt < 4; ++mt) {
        // A 16x32 layout: lane(m,hf): e[0..7]=A[m][8hf+j], e[8..15]=A[m][16+8hf+j]
        BfOp a;
        a.q[0] = *reinterpret_cast<const uint4*>(&a_s[mt * 16 + m][hf * 8]);
        a.q[1] = *reinterpret_cast<const uint4*>(&a_s[mt * 16 + m][16 + hf * 8]);
        v8f* cc2 = (mt == 0) ? &c0 : (mt == 1) ? &c1 : (mt == 2) ? &c2 : &c3;
        *cc2 = __builtin_amdgcn_wmma_f32_16x16x32_bf16(false, a.v, false, b.v,
                                                       (short)0, *cc2, false, false);
      }
    }
    __syncthreads();
  }
  // epilogue: D layout row = r + 8*(lane>=16), col = lane%16 within wave tile
  const int m = lane & 15, hf = lane >> 4;
  const int col = nbase + (wave << 4) + m;
  const float bv = bias[col];
#pragma unroll
  for (int mt = 0; mt < 4; ++mt) {
    const v8f* cc2 = (mt == 0) ? &c0 : (mt == 1) ? &c1 : (mt == 2) ? &c2 : &c3;
#pragma unroll
    for (int r = 0; r < 8; ++r) {
      const int row = mbase + mt * 16 + r + (hf << 3);
      const float val = (*cc2)[r] + bv;
      if (mode == 0) {
        out0[(size_t)row * Nc + col] = val;
      } else {
        const int which = col >> 10, rem = col & 1023;
        const int hh = rem >> 6, dd = rem & 63;
        float* dst = (which == 0) ? out0 : (which == 1) ? out1 : out2;
        dst[(size_t)((hh << 10) + row) * 64 + dd] = val;
      }
    }
  }
}

// ==========================================================================
// KV prep: one-pass f32->bf16 of the concatenated 5120-key K/V streams.
//   kbf [h][t][d]   row-major bf16        (WMMA A operand source)
//   vtbf[h][d][t]   transposed bf16       (WMMA B operand source, via LDS)
// grid = (TK/64, H), block = 256. Each element converted exactly once
// (attention re-reads bf16 from L2, not f32, and does zero conversion VALU).
// ==========================================================================
__global__ __launch_bounds__(256) void kvprep_kernel(
    const float* __restrict__ past_k, const float* __restrict__ past_v,
    const float* __restrict__ k_new,  const float* __restrict__ v_new,
    unsigned short* __restrict__ kbf, unsigned short* __restrict__ vtbf)
{
  __shared__ __align__(16) unsigned short vt[64][LDSP];
  const int h  = blockIdx.y, t0 = blockIdx.x * 64;
  const int tid = threadIdx.x;
  const int row = tid >> 2, seg = (tid & 3) << 4;  // 64 rows x 4 chunks of 16
  const int tg = t0 + row;
  const float* ks;
  const float* vs;
  if (tg < TPAST_) {
    ks = &past_k[((size_t)h * TPAST_ + tg) * 64];
    vs = &past_v[((size_t)h * TPAST_ + tg) * 64];
  } else {
    ks = &k_new[((size_t)(h << 10) + (tg - TPAST_)) * 64];
    vs = &v_new[((size_t)(h << 10) + (tg - TPAST_)) * 64];
  }
  float4 kv[4], vv[4];
#pragma unroll
  for (int j4 = 0; j4 < 4; ++j4) {
    kv[j4] = *reinterpret_cast<const float4*>(&ks[seg + j4 * 4]);
    vv[j4] = *reinterpret_cast<const float4*>(&vs[seg + j4 * 4]);
  }
  // K: convert and store row-major directly (32B per thread, coalesced)
  unsigned kk[8];
#pragma unroll
  for (int j4 = 0; j4 < 4; ++j4) {
    kk[2 * j4]     = pk2(kv[j4].x, kv[j4].y);
    kk[2 * j4 + 1] = pk2(kv[j4].z, kv[j4].w);
  }
  unsigned short* ko = &kbf[((size_t)h * TK_ + tg) * 64 + seg];
  *reinterpret_cast<uint4*>(ko)     = *reinterpret_cast<uint4*>(&kk[0]);
  *reinterpret_cast<uint4*>(ko + 8) = *reinterpret_cast<uint4*>(&kk[4]);
  // V: transpose through LDS
#pragma unroll
  for (int j4 = 0; j4 < 4; ++j4) {
    vt[seg + j4 * 4 + 0][row] = f2bf(vv[j4].x);
    vt[seg + j4 * 4 + 1][row] = f2bf(vv[j4].y);
    vt[seg + j4 * 4 + 2][row] = f2bf(vv[j4].z);
    vt[seg + j4 * 4 + 3][row] = f2bf(vv[j4].w);
  }
  __syncthreads();
  // write V^T rows coalesced: thread covers (d = row, keys seg..seg+15)
  const uint4 a = *reinterpret_cast<const uint4*>(&vt[row][seg]);
  const uint4 b = *reinterpret_cast<const uint4*>(&vt[row][seg + 8]);
  unsigned short* vo = &vtbf[((size_t)h * 64 + row) * TK_ + t0 + seg];
  *reinterpret_cast<uint4*>(vo)     = a;
  *reinterpret_cast<uint4*>(vo + 8) = b;
}

// ==========================================================================
// Flash attention. grid = (N/64 q-blocks, H heads), block = 128 (4 waves).
// Each wave owns 16 queries. S^T = K*Q^T form: each lane holds one query's
// scores (softmax = in-register + one xor-16 shuffle); exp(P) repacks
// lane-for-lane into the PV A operand. K/V staged bf16 via double-buffered
// async global->LDS copies (ASYNCcnt) when available.
// ==========================================================================
__global__ __launch_bounds__(128) void attn_kernel(
    const float* __restrict__ q_buf,
    const unsigned short* __restrict__ kbf,
    const unsigned short* __restrict__ vtbf,
    float* __restrict__ o_buf)
{
  __shared__ __align__(16) unsigned short k_s[2][32][LDSP];   // keys x d
  __shared__ __align__(16) unsigned short vT_s[2][64][LDSP];  // d x keys
  const int tid  = threadIdx.x;
  const int lane = tid & 31, wave = tid >> 5;
  const int h = blockIdx.y;
  const int qbase = blockIdx.x * 64 + wave * 16;
  const int m = lane & 15, hf = lane >> 4;
  const float qscale = 0.125f;  // D^-0.5, folded into Q

  // staging geometry (pure bf16 copies, no conversion)
  const int krow = tid >> 2, kseg = (tid & 3) << 4;  // 32 x 64
  const int vrow = tid >> 1, vseg = (tid & 1) << 4;  // 64 x 32
  const unsigned short* kgb = &kbf[((size_t)h * TK_) * 64 + (size_t)krow * 64 + kseg];
  const unsigned short* vgb = &vtbf[((size_t)h * 64 + vrow) * TK_ + vseg];
  auto stage = [&](int kb, int buf) {
    const unsigned short* kg = kgb + (size_t)(kb << 5) * 64;
    const unsigned short* vg = vgb + (kb << 5);
    cp16_lds(kg,     &k_s[buf][krow][kseg]);
    cp16_lds(kg + 8, &k_s[buf][krow][kseg + 8]);
    cp16_lds(vg,     &vT_s[buf][vrow][vseg]);
    cp16_lds(vg + 8, &vT_s[buf][vrow][vseg + 8]);
  };

  // Q^T as two B operands (d-chunks [0,32) and [32,64)), scaled by 1/sqrt(D)
  BfOp bq0, bq1;
  {
    const float* qrow = &q_buf[(size_t)((h << 10) + (qbase + m)) * 64];
    float4 qv[8];
#pragma unroll
    for (int i = 0; i < 4; ++i) {
      qv[i]     = *reinterpret_cast<const float4*>(&qrow[hf * 8 + i * 4 + (i >> 1) * 8]);
      qv[4 + i] = *reinterpret_cast<const float4*>(&qrow[32 + hf * 8 + i * 4 + (i >> 1) * 8]);
    }
#pragma unroll
    for (int i = 0; i < 4; ++i) {
      bq0.u[2 * i]     = pk2(qv[i].x * qscale, qv[i].y * qscale);
      bq0.u[2 * i + 1] = pk2(qv[i].z * qscale, qv[i].w * qscale);
      bq1.u[2 * i]     = pk2(qv[4 + i].x * qscale, qv[4 + i].y * qscale);
      bq1.u[2 * i + 1] = pk2(qv[4 + i].z * qscale, qv[4 + i].w * qscale);
    }
  }

  v8f o0 = {}, o1 = {}, o2 = {}, o3 = {};
  float mrun = -1.0e30f, lsum = 0.f;

  stage(0, 0);
#if USE_ASYNC_LDS
  __builtin_amdgcn_s_wait_asynccnt(0);
#endif
  __syncthreads();

  int cur = 0;
  const int NKB = TK_ / 32;
  for (int kb = 0; kb < NKB; ++kb) {
    // prefetch next key-block into the other LDS buffer (overlaps WMMAs)
    if (kb + 1 < NKB) stage(kb + 1, cur ^ 1);

    // ---- S^T tiles: two 16-key subtiles, each = 2 WMMAs over d=64 ----
    v8f c0 = {}, c1 = {};
#pragma unroll
    for (int s = 0; s < 2; ++s) {
      BfOp a0, a1;  // A = K rows (key-major), d-chunks 0..31 / 32..63
      a0.q[0] = *reinterpret_cast<const uint4*>(&k_s[cur][s * 16 + m][hf * 8]);
      a0.q[1] = *reinterpret_cast<const uint4*>(&k_s[cur][s * 16 + m][16 + hf * 8]);
      a1.q[0] = *reinterpret_cast<const uint4*>(&k_s[cur][s * 16 + m][32 + hf * 8]);
      a1.q[1] = *reinterpret_cast<const uint4*>(&k_s[cur][s * 16 + m][48 + hf * 8]);
      v8f cc = {};
      cc = __builtin_amdgcn_wmma_f32_16x16x32_bf16(false, a0.v, false, bq0.v, (short)0, cc, false, false);
      cc = __builtin_amdgcn_wmma_f32_16x16x32_bf16(false, a1.v, false, bq1.v, (short)0, cc, false, false);
      if (s == 0) c0 = cc; else c1 = cc;
    }

    // ---- online softmax: lane owns query q=m, keys {8hf+r} U {16+8hf+r} ----
    float bmax = c0[0];
#pragma unroll
    for (int r = 0; r < 8; ++r) { bmax = fmaxf(bmax, c0[r]); bmax = fmaxf(bmax, c1[r]); }
    bmax = fmaxf(bmax, __shfl_xor(bmax, 16, 32));
    const float mnew  = fmaxf(mrun, bmax);
    const float alpha = __expf(mrun - mnew);
    float p0[8], p1[8], part = 0.f;
#pragma unroll
    for (int r = 0; r < 8; ++r) {
      p0[r] = __expf(c0[r] - mnew);
      p1[r] = __expf(c1[r] - mnew);
      part += p0[r] + p1[r];
    }
    part += __shfl_xor(part, 16, 32);
    lsum = lsum * alpha + part;
    mrun = mnew;
    // rescale O accumulators (O rows are queries r+8hf; fetch that query's alpha)
#pragma unroll
    for (int r = 0; r < 8; ++r) {
      const float fac = __shfl(alpha, r + (hf << 3), 32);
      o0[r] *= fac; o1[r] *= fac; o2[r] *= fac; o3[r] *= fac;
    }

    // ---- PV: A = P (lane-local repack, no LDS), B = V^T rows ----
    BfOp pa;
#pragma unroll
    for (int i = 0; i < 4; ++i) {
      pa.u[i]     = pk2(p0[2 * i], p0[2 * i + 1]);   // keys 8hf+2i,+1
      pa.u[4 + i] = pk2(p1[2 * i], p1[2 * i + 1]);   // keys 16+8hf+2i,+1
    }
#pragma unroll
    for (int dc = 0; dc < 4; ++dc) {
      BfOp bv;  // lane(n=d,hf): e[0..7] = V[8hf+j][dc*16+n], e[8..15] = V[16+8hf+j][dc*16+n]
      bv.q[0] = *reinterpret_cast<const uint4*>(&vT_s[cur][dc * 16 + m][hf * 8]);
      bv.q[1] = *reinterpret_cast<const uint4*>(&vT_s[cur][dc * 16 + m][16 + hf * 8]);
      v8f* oo = (dc == 0) ? &o0 : (dc == 1) ? &o1 : (dc == 2) ? &o2 : &o3;
      *oo = __builtin_amdgcn_wmma_f32_16x16x32_bf16(false, pa.v, false, bv.v,
                                                    (short)0, *oo, false, false);
    }

#if USE_ASYNC_LDS
    __builtin_amdgcn_s_wait_asynccnt(0);
#endif
    __syncthreads();
    cur ^= 1;
  }

  // ---- normalize and store o in (N, H*D) layout for the proj GEMM ----
  const float linv = 1.f / lsum;
#pragma unroll
  for (int r = 0; r < 8; ++r) {
    const float li  = __shfl(linv, r + (hf << 3), 32);
    const int   row = qbase + r + (hf << 3);
    float* op = &o_buf[(size_t)row * DIM_ + h * 64];
    op[0 * 16 + m] = o0[r] * li;
    op[1 * 16 + m] = o1[r] * li;
    op[2 * 16 + m] = o2[r] * li;
    op[3 * 16 + m] = o3[r] * li;
  }
}

// ==========================================================================
// Pruning step 1: qbar[d] = (1/(H*69)) * sum_h sum_t w(t) q[h][t][d]
// ==========================================================================
__global__ __launch_bounds__(256) void qbar_kernel(const float* __restrict__ q_buf,
                                                   float* __restrict__ qbar) {
  __shared__ float part[4][64];
  const int d = threadIdx.x & 63;
  const int p = threadIdx.x >> 6;
  float acc = 0.f;
  for (int hh = 0; hh < H_; ++hh) {
    const float* qh = &q_buf[(size_t)(hh << 10) * 64];
    for (int t = p * 256; t < p * 256 + 256; ++t) {
      const float w = (t < 5) ? 1.f : (t < 1013) ? (1.f / 16.f) : (1.f / 11.f);
      acc += w * qh[(size_t)t * 64 + d];
    }
  }
  part[p][d] = acc;
  __syncthreads();
  if (threadIdx.x < 64)
    qbar[d] = (part[0][d] + part[1][d] + part[2][d] + part[3][d]) * (1.f / (16.f * 69.f));
}

// Pruning step 2: scores[p] = (1/H) sum_h <qbar, past_k[h][1024+p]>
__global__ __launch_bounds__(256) void score_kernel(const float* __restrict__ qbar,
                                                    const float* __restrict__ past_k,
                                                    float* __restrict__ scores) {
  __shared__ float qb[64];
  if (threadIdx.x < 64) qb[threadIdx.x] = qbar[threadIdx.x];
  __syncthreads();
  const int p = blockIdx.x * 256 + threadIdx.x;
  if (p >= MID_) return;
  float s = 0.f;
  for (int hh = 0; hh < H_; ++hh) {
    const float* kr = &past_k[((size_t)hh * TPAST_ + 1024 + p) * 64];
#pragma unroll
    for (int dd = 0; dd < 64; ++dd) s += qb[dd] * kr[dd];
  }
  scores[p] = s * (1.f / 16.f);
}

// Pruning step 3: rank-select top 2048 of 3072 (top_k tie-break: lower index
// wins), emit sorted keep[4096] = [0,1024) U mid U [4096,5120)
__global__ __launch_bounds__(256) void select_kernel(const float* __restrict__ scores,
                                                     int* __restrict__ keep) {
  __shared__ float s[MID_];
  __shared__ unsigned char kept[MID_];
  for (int i = threadIdx.x; i < MID_; i += 256) s[i] = scores[i];
  __syncthreads();
  for (int i = threadIdx.x; i < MID_; i += 256) {
    const float si = s[i];
    int rank = 0;
    for (int j = 0; j < MID_; ++j)
      rank += (s[j] > si) ? 1 : ((s[j] == si && j < i) ? 1 : 0);
    kept[i] = (rank < KEEPMID_) ? 1 : 0;
  }
  __syncthreads();
  for (int i = threadIdx.x; i < 1024; i += 256) {
    keep[i] = i;
    keep[1024 + KEEPMID_ + i] = 4096 + i;
  }
  for (int i = threadIdx.x; i < MID_; i += 256) {
    if (kept[i]) {
      int pos = 0;
      for (int j = 0; j < i; ++j) pos += kept[j];
      keep[1024 + pos] = 1024 + i;
    }
  }
}

// Pruning step 4: gather new_k/new_v (float4 granularity, both in one grid)
__global__ __launch_bounds__(256) void gather_kernel(
    const int* __restrict__ keep,
    const float* __restrict__ past_k, const float* __restrict__ past_v,
    const float* __restrict__ k_new,  const float* __restrict__ v_new,
    float* __restrict__ out_k, float* __restrict__ out_v)
{
  const long gid = (long)blockIdx.x * 256 + threadIdx.x;  // 2 * 16*4096*16 float4
  const int sel = (int)(gid >> 20);
  const int r   = (int)(gid & ((1 << 20) - 1));
  const int hh  = r >> 16;
  const int rr  = r & 65535;
  const int j   = rr >> 4;
  const int c4  = rr & 15;
  const int t   = keep[j];
  const float* past = sel ? past_v : past_k;
  const float* nw   = sel ? v_new : k_new;
  float* dst        = sel ? out_v : out_k;
  const float4 val = (t < TPAST_)
      ? *reinterpret_cast<const float4*>(&past[((size_t)hh * TPAST_ + t) * 64 + c4 * 4])
      : *reinterpret_cast<const float4*>(&nw[((size_t)(hh << 10) + (t - TPAST_)) * 64 + c4 * 4]);
  *reinterpret_cast<float4*>(&dst[((size_t)hh * 4096 + j) * 64 + c4 * 4]) = val;
}

// ==========================================================================
extern "C" void kernel_launch(void* const* d_in, const int* in_sizes, int n_in,
                              void* d_out, int out_size, void* d_ws, size_t ws_size,
                              hipStream_t stream) {
  (void)in_sizes; (void)n_in; (void)out_size; (void)ws_size;
  const float* x      = (const float*)d_in[0];
  const float* past_k = (const float*)d_in[1];
  const float* past_v = (const float*)d_in[2];
  const float* qkv_w  = (const float*)d_in[3];
  const float* qkv_b  = (const float*)d_in[4];
  const float* proj_w = (const float*)d_in[5];
  const float* proj_b = (const float*)d_in[6];

  float* out   = (float*)d_out;                 // (1024,1024)
  float* out_k = out + 1024 * 1024;             // (16,4096,64)
  float* out_v = out_k + 16 * 4096 * 64;        // (16,4096,64)

  float* ws     = (float*)d_ws;                 // ~38 MB used
  float* q_buf  = ws;                           // [16][1024][64] f32
  float* k_new  = q_buf + (1 << 20);
  float* v_new  = k_new + (1 << 20);
  float* o_buf  = v_new + (1 << 20);            // [1024][1024] f32
  float* qbar   = o_buf + (1 << 20);            // [64]
  float* scores = qbar + 64;                    // [3072]
  int*   keep   = (int*)(scores + MID_);        // [4096]
  unsigned short* kbf  = (unsigned short*)(ws + (4u << 20) + 8192);  // [16][5120][64] bf16
  unsigned short* vtbf = kbf + (size_t)H_ * TK_ * 64;                // [16][64][5120] bf16

  // QKV GEMM with head-major scatter epilogue
  gemm_bf16_kernel<<<dim3(3072 / 64, 1024 / 64), 128, 0, stream>>>(
      x, qkv_w, qkv_b, 1024, 1024, 3072, 1, q_buf, k_new, v_new);
  // One-pass bf16 conversion (+V transpose) of the 5120-key K/V streams
  kvprep_kernel<<<dim3(TK_ / 64, H_), 256, 0, stream>>>(
      past_k, past_v, k_new, v_new, kbf, vtbf);
  // Flash attention over 5120 keys
  attn_kernel<<<dim3(16, 16), 128, 0, stream>>>(q_buf, kbf, vtbf, o_buf);
  // Output projection
  gemm_bf16_kernel<<<dim3(1024 / 64, 1024 / 64), 128, 0, stream>>>(
      o_buf, proj_w, proj_b, 1024, 1024, 1024, 0, out, nullptr, nullptr);
  // KV-cache pruning
  qbar_kernel<<<1, 256, 0, stream>>>(q_buf, qbar);
  score_kernel<<<dim3(12), 256, 0, stream>>>(qbar, past_k, scores);
  select_kernel<<<1, 256, 0, stream>>>(scores, keep);
  gather_kernel<<<dim3(8192), 256, 0, stream>>>(keep, past_k, past_v, k_new, v_new, out_k, out_v);
}
